// FNetTransformer_22952305229872
// MI455X (gfx1250) — matrix operand order repaired and checked by reading, not compiled, not run on hardware
//
#include <hip/hip_runtime.h>
#include <math.h>

// Problem dims (fixed by the reference)
#define DIMD   1024
#define SEQ    2048
#define BATCH  8
#define NROWS  (BATCH * SEQ)      // 16384 token rows
#define LN_EPS 1e-5f

typedef __attribute__((ext_vector_type(2)))  float  v2f;
typedef __attribute__((ext_vector_type(8)))  float  v8f;
typedef __attribute__((ext_vector_type(16))) __bf16 v16bf;

union BF16x16 { v16bf v; unsigned int u[8]; };

// ---------------- workspace layout (bytes) ----------------
#define OFF_C    ((size_t)0)
#define OFF_W1T  (OFF_C   + (size_t)SEQ * SEQ * 4)
#define OFF_W2T  (OFF_W1T + (size_t)DIMD * DIMD * 2)
#define OFF_Y    (OFF_W2T + (size_t)DIMD * DIMD * 2)
#define OFF_T    (OFF_Y   + (size_t)NROWS * DIMD * 4)
#define OFF_H    (OFF_T   + (size_t)NROWS * DIMD * 2)

// ---------------- prep kernels ----------------
__global__ __launch_bounds__(256) void build_cos_kernel(float* __restrict__ Cm) {
    int i  = blockIdx.x * 256 + threadIdx.x;     // [0, S*S)
    int s  = i >> 11;
    int sp = i & (SEQ - 1);
    int prod = (s * sp) & (SEQ - 1);             // (s*s') mod S, S = 2^11
    Cm[i] = cosf((2.0f * 3.14159265358979f / (float)SEQ) * (float)prod);
}

__global__ __launch_bounds__(256) void wtrans_bf16_kernel(const float* __restrict__ W,
                                                          __bf16* __restrict__ Wt) {
    int i = blockIdx.x * 256 + threadIdx.x;      // [0, D*D)
    int n = i >> 10;
    int k = i & (DIMD - 1);
    Wt[(size_t)n * DIMD + k] = (__bf16)W[(size_t)k * DIMD + n];
}

// ---------------- DFT-as-GEMM (fp32 WMMA) + residual add ----------------
// G[b,s,e] = sum_{s'} C[s,s'] x[b,s',e];  y[b,s,(-e)%D] = x[..] + D*G[b,s,e]
// Block tile 64(M) x 128(N); 8 waves in 2x4 grid; wave tile 32x32 (2x2 WMMA tiles).
// Double-buffered LDS: one barrier per 32-K chunk; stage(t+1) overlaps compute(t).

__device__ __forceinline__ void dft_stage(const float* __restrict__ Cm,
                                          const float* __restrict__ xb,
                                          int s0, int e0, int kk, int tid,
                                          float* __restrict__ la,
                                          float* __restrict__ lb) {
#pragma unroll
    for (int i = 0; i < 4; ++i) {               // A: C[s0..+63][kk..+31]
        int idx = tid + i * 256;
        int r   = idx >> 4;
        int c2  = (idx & 15) * 2;
        float2 t2 = *(const float2*)(Cm + (size_t)(s0 + r) * SEQ + kk + c2);
        la[r * 33 + c2]     = t2.x;
        la[r * 33 + c2 + 1] = t2.y;
    }
#pragma unroll
    for (int i = 0; i < 4; ++i) {               // B: x[kk..+31][e0..+127]
        int idx = tid + i * 256;
        int rr  = idx >> 5;
        int c4  = idx & 31;
        *(float4*)&lb[rr * 128 + c4 * 4] =
            *(const float4*)(xb + (size_t)(kk + rr) * DIMD + e0 + c4 * 4);
    }
}

__global__ __launch_bounds__(256) void dft_add_kernel(const float* __restrict__ x,
                                                      const float* __restrict__ Cm,
                                                      float* __restrict__ y) {
    __shared__ float lds_a[2][64 * 33];
    __shared__ float lds_b[2][32 * 128];

    const int tid  = threadIdx.x;
    const int wv   = tid >> 5;
    const int ln   = tid & 31;
    const int m    = ln & 15;
    const int half = ln >> 4;
    const int n    = ln & 15;
    const int wm   = wv >> 2;
    const int wn   = wv & 3;

    const int s0 = blockIdx.x * 64;
    const int e0 = blockIdx.y * 128;
    const float* xb = x + (size_t)blockIdx.z * SEQ * DIMD;

    v8f c[2][2] = {};

    dft_stage(Cm, xb, s0, e0, 0, tid, lds_a[0], lds_b[0]);
    __syncthreads();

    const int NT = SEQ / 32;
    for (int t = 0; t < NT; ++t) {
        const int cur = t & 1;
        if (t + 1 < NT)
            dft_stage(Cm, xb, s0, e0, (t + 1) * 32, tid,
                      lds_a[cur ^ 1], lds_b[cur ^ 1]);

        const float* la = lds_a[cur];
        const float* lb = lds_b[cur];
#pragma unroll
        for (int k = 0; k < 32; k += 4) {
            v2f a[2], b[2];
#pragma unroll
            for (int i = 0; i < 2; ++i) {
                const float* ap = &la[(wm * 32 + i * 16 + m) * 33 + k + half * 2];
                a[i].x = ap[0];
                a[i].y = ap[1];
            }
#pragma unroll
            for (int j = 0; j < 2; ++j) {
                int col = wn * 32 + j * 16 + n;
                b[j].x = lb[(k + half * 2)     * 128 + col];
                b[j].y = lb[(k + half * 2 + 1) * 128 + col];
            }
#pragma unroll
            for (int i = 0; i < 2; ++i)
#pragma unroll
                for (int j = 0; j < 2; ++j)
                    c[i][j] = __builtin_amdgcn_wmma_f32_16x16x4_f32(
                        false, a[i], false, b[j], (short)0, c[i][j], false, false);
        }
        __syncthreads();
    }

    // epilogue: hidden-axis reversal + residual add, scale by D
#pragma unroll
    for (int j = 0; j < 2; ++j) {
        int e = e0 + wn * 32 + j * 16 + n;
        int d = (DIMD - e) & (DIMD - 1);
#pragma unroll
        for (int i = 0; i < 2; ++i) {
#pragma unroll
            for (int v = 0; v < 8; ++v) {
                int row = s0 + wm * 32 + i * 16 + half * 8 + v;
                size_t off = (size_t)blockIdx.z * SEQ * DIMD + (size_t)row * DIMD + d;
                y[off] = x[off] + 1024.0f * c[i][j][v];
            }
        }
    }
}

// ---------------- LayerNorm (block per row) ----------------
__global__ __launch_bounds__(256) void ln_kernel(const float* __restrict__ y,
                                                 const float* __restrict__ g,
                                                 const float* __restrict__ bb,
                                                 __bf16* __restrict__ t) {
    __shared__ float red[256];
    const int r = blockIdx.x;
    const float* yr = y + (size_t)r * DIMD;

    float vals[4], s = 0.f;
#pragma unroll
    for (int i = 0; i < 4; ++i) { vals[i] = yr[threadIdx.x + i * 256]; s += vals[i]; }
    red[threadIdx.x] = s; __syncthreads();
    for (int off = 128; off > 0; off >>= 1) {
        if (threadIdx.x < off) red[threadIdx.x] += red[threadIdx.x + off];
        __syncthreads();
    }
    float mu = red[0] * (1.0f / DIMD);
    __syncthreads();

    float vs = 0.f;
#pragma unroll
    for (int i = 0; i < 4; ++i) { float d = vals[i] - mu; vs += d * d; }
    red[threadIdx.x] = vs; __syncthreads();
    for (int off = 128; off > 0; off >>= 1) {
        if (threadIdx.x < off) red[threadIdx.x] += red[threadIdx.x + off];
        __syncthreads();
    }
    float inv = rsqrtf(red[0] * (1.0f / DIMD) + LN_EPS);

#pragma unroll
    for (int i = 0; i < 4; ++i) {
        int cidx = threadIdx.x + i * 256;
        t[(size_t)r * DIMD + cidx] = (__bf16)((vals[i] - mu) * inv * g[cidx] + bb[cidx]);
    }
}

// ---------------- FFN GEMM (bf16 WMMA) ----------------
// Block tile 64(M) x 128(N); 8 waves in 2x4; wave tile 32x32 (2x2 WMMA tiles).
// Double-buffered LDS; mode 0: bias+GELU -> bf16, mode 1: bias * mask -> f32.

__device__ __forceinline__ void ffn_stage(const __bf16* __restrict__ Abf,
                                          const __bf16* __restrict__ Bt,
                                          int m0, int n0, int kk, int tid,
                                          unsigned int* __restrict__ la,
                                          unsigned int* __restrict__ lb) {
#pragma unroll
    for (int i = 0; i < 4; ++i) {               // A: 64 rows x 16 dwords
        int idx = tid + i * 256;
        int r   = idx >> 4;
        int cd  = idx & 15;
        la[r * 17 + cd] =
            *(const unsigned int*)(Abf + (size_t)(m0 + r) * DIMD + kk + cd * 2);
    }
#pragma unroll
    for (int i = 0; i < 2; ++i) {               // B: 128 cols x 16 dwords
        int idx = tid + i * 256;                // uint4 index
        int col = idx >> 2;
        int q   = idx & 3;
        uint4 w = *(const uint4*)(Bt + (size_t)(n0 + col) * DIMD + kk + q * 8);
        *(uint4*)&lb[col * 20 + q * 4] = w;
    }
}

__global__ __launch_bounds__(256) void ffn_gemm_kernel(const __bf16* __restrict__ Abf,  // M x D
                                                       const __bf16* __restrict__ Bt,   // N x D (transposed)
                                                       const float* __restrict__ bias,
                                                       const float* __restrict__ mask,
                                                       __bf16* __restrict__ outbf,
                                                       float* __restrict__ outf,
                                                       int mode) {
    __shared__ unsigned int lds_a[2][64 * 17];
    __shared__ unsigned int lds_b[2][128 * 20];

    const int tid  = threadIdx.x;
    const int wv   = tid >> 5;
    const int ln   = tid & 31;
    const int m    = ln & 15;
    const int half = ln >> 4;
    const int n    = ln & 15;
    const int wm   = wv >> 2;
    const int wn   = wv & 3;

    const int m0 = blockIdx.x * 64;
    const int n0 = blockIdx.y * 128;

    v8f c[2][2] = {};

    ffn_stage(Abf, Bt, m0, n0, 0, tid, lds_a[0], lds_b[0]);
    __syncthreads();

    const int NT = DIMD / 32;
    for (int t = 0; t < NT; ++t) {
        const int cur = t & 1;
        if (t + 1 < NT)
            ffn_stage(Abf, Bt, m0, n0, (t + 1) * 32, tid,
                      lds_a[cur ^ 1], lds_b[cur ^ 1]);

        const unsigned int* la = lds_a[cur];
        const unsigned int* lb = lds_b[cur];

        BF16x16 a[2], b[2];
#pragma unroll
        for (int i = 0; i < 2; ++i) {
            int row = wm * 32 + i * 16 + m;
#pragma unroll
            for (int j = 0; j < 4; ++j) a[i].u[j]     = la[row * 17 + half * 4 + j];
#pragma unroll
            for (int j = 0; j < 4; ++j) a[i].u[4 + j] = la[row * 17 + 8 + half * 4 + j];
        }
#pragma unroll
        for (int j = 0; j < 2; ++j) {
            int col = wn * 32 + j * 16 + n;
            uint4 q0 = *(const uint4*)&lb[col * 20 + half * 4];
            uint4 q1 = *(const uint4*)&lb[col * 20 + 8 + half * 4];
            b[j].u[0] = q0.x; b[j].u[1] = q0.y; b[j].u[2] = q0.z; b[j].u[3] = q0.w;
            b[j].u[4] = q1.x; b[j].u[5] = q1.y; b[j].u[6] = q1.z; b[j].u[7] = q1.w;
        }
#pragma unroll
        for (int i = 0; i < 2; ++i)
#pragma unroll
            for (int j = 0; j < 2; ++j)
                c[i][j] = __builtin_amdgcn_wmma_f32_16x16x32_bf16(
                    false, a[i].v, false, b[j].v, (short)0, c[i][j], false, false);
        __syncthreads();
    }

#pragma unroll
    for (int j = 0; j < 2; ++j) {
        int col = n0 + wn * 32 + j * 16 + n;
        float bcol = bias[col];
#pragma unroll
        for (int i = 0; i < 2; ++i) {
#pragma unroll
            for (int v = 0; v < 8; ++v) {
                int row   = m0 + wm * 32 + i * 16 + half * 8 + v;
                float val = c[i][j][v] + bcol;
                if (mode == 0) {
                    float gl = 0.5f * val * (1.0f + erff(val * 0.70710678118654752f));
                    outbf[(size_t)row * DIMD + col] = (__bf16)gl;
                } else {
                    outf[(size_t)row * DIMD + col] = val * mask[row];
                }
            }
        }
    }
}

// ---------------- host launcher ----------------
extern "C" void kernel_launch(void* const* d_in, const int* in_sizes, int n_in,
                              void* d_out, int out_size, void* d_ws, size_t ws_size,
                              hipStream_t stream) {
    const float* x    = (const float*)d_in[0];
    const float* mask = (const float*)d_in[1];
    const float* ln_g = (const float*)d_in[2];
    const float* ln_b = (const float*)d_in[3];
    const float* w1   = (const float*)d_in[4];
    const float* b1   = (const float*)d_in[5];
    const float* w2   = (const float*)d_in[6];
    const float* b2   = (const float*)d_in[7];
    float* out = (float*)d_out;

    char* ws = (char*)d_ws;
    float*  Cm  = (float*) (ws + OFF_C);
    __bf16* w1t = (__bf16*)(ws + OFF_W1T);
    __bf16* w2t = (__bf16*)(ws + OFF_W2T);
    float*  y   = (float*) (ws + OFF_Y);
    __bf16* tbf = (__bf16*)(ws + OFF_T);
    __bf16* hbf = (__bf16*)(ws + OFF_H);

    build_cos_kernel<<<(SEQ * SEQ) / 256, 256, 0, stream>>>(Cm);
    wtrans_bf16_kernel<<<(DIMD * DIMD) / 256, 256, 0, stream>>>(w1, w1t);
    wtrans_bf16_kernel<<<(DIMD * DIMD) / 256, 256, 0, stream>>>(w2, w2t);

    dft_add_kernel<<<dim3(SEQ / 64, DIMD / 128, BATCH), 256, 0, stream>>>(x, Cm, y);

    ln_kernel<<<NROWS, 256, 0, stream>>>(y, ln_g, ln_b, tbf);

    ffn_gemm_kernel<<<dim3(NROWS / 64, DIMD / 128), 256, 0, stream>>>(
        tbf, w1t, b1, nullptr, hbf, nullptr, 0);
    ffn_gemm_kernel<<<dim3(NROWS / 64, DIMD / 128), 256, 0, stream>>>(
        hbf, w2t, b2, mask, nullptr, out, 1);
}
